// AttentionGatedDeltaNet_43645457662583
// MI455X (gfx1250) — matrix operand order repaired and checked by reading, not compiled, not run on hardware
//
#include <hip/hip_runtime.h>
#include <hip/hip_bf16.h>

// ---------------- problem constants (match reference) ----------------
#define T_      512
#define D_      2048
#define HK_     16
#define HV_     32
#define DK_     128
#define DV_     128
#define KEY_DIM_   2048        // HK*DK
#define VAL_DIM_   4096        // HV*DV
#define CONV_DIM_  8192        // 2*KEY_DIM + VAL_DIM
#define EPS_    1e-6f

#define KSLAB_  64             // K-depth staged per TDM transfer (2 WMMA steps)
#define LROW_   68             // LDS row stride in floats (64 + 4 pad -> bank rotate)

typedef __attribute__((ext_vector_type(16))) __bf16 v16bf;
typedef __attribute__((ext_vector_type(8)))  float  v8f;
typedef __attribute__((ext_vector_type(4)))  unsigned int u32x4;
typedef __attribute__((ext_vector_type(8)))  int          i32x8;
typedef __attribute__((ext_vector_type(4)))  int          i32x4;

__device__ __forceinline__ float silu_(float x)    { return x / (1.f + __expf(-x)); }
__device__ __forceinline__ float sigmoid_(float x) { return 1.f / (1.f + __expf(-x)); }

// =====================================================================
// Tensor Data Mover: async 2D tile load Global -> LDS (ISA §8).
// Tile: tile_d1 rows x tile_d0 elems (f32). LDS padded 4 dwords every
// 64 dwords (pad_interval=5, pad_amount=3) => row stride LROW_ floats.
// =====================================================================
__device__ __forceinline__ void tdm_load_2d(unsigned lds_addr, const void* gp,
                                            unsigned tile_d0, unsigned tile_d1,
                                            unsigned tensor_d0, unsigned tensor_d1,
                                            unsigned long long stride0)
{
    const unsigned long long ga = (unsigned long long)(uintptr_t)gp;
    u32x4 g0;
    g0[0] = 1u;                                            // count=1, user D#
    g0[1] = lds_addr;                                      // lds_addr (bytes)
    g0[2] = (unsigned)(ga & 0xFFFFFFFFu);                  // global_addr[31:0]
    g0[3] = (unsigned)((ga >> 32) & 0x01FFFFFFu)           // global_addr[56:32]
          | (2u << 30);                                    // type = 2 (image)
    i32x8 g1;
    g1[0] = (int)((2u << 16)                               // data_size = 4 bytes
          | (1u << 20)                                     // pad_enable
          | (5u << 22)                                     // pad_interval: 64 dw
          | (3u << 25));                                   // pad_amount: 4 dw
    g1[1] = (int)((tensor_d0 & 0xFFFFu) << 16);            // tensor_dim0[15:0]
    g1[2] = (int)(((tensor_d0 >> 16) & 0xFFFFu)
          | ((tensor_d1 & 0xFFFFu) << 16));                // td0[31:16] | td1[15:0]
    g1[3] = (int)(((tensor_d1 >> 16) & 0xFFFFu)
          | (tile_d0 << 16));                              // td1[31:16] | tile_dim0
    g1[4] = (int)(tile_d1 & 0xFFFFu);                      // tile_dim1 (tile_dim2=0)
    g1[5] = (int)(stride0 & 0xFFFFFFFFull);                // tensor_dim0_stride lo
    g1[6] = (int)((stride0 >> 32) & 0xFFFFull);            // stride hi (dim1_stride=0)
    g1[7] = 0;
    const i32x4 z4 = {};
#if defined(__clang_major__) && (__clang_major__ >= 23)
    const i32x8 z8 = {};
    __builtin_amdgcn_tensor_load_to_lds(g0, g1, z4, z4, z8, 0);
#else
    __builtin_amdgcn_tensor_load_to_lds(g0, g1, z4, z4, 0);
#endif
}

// =====================================================================
// Kernel 1: Y[M,N] = X[M,K] @ W[N,K]^T.
// Block = 8 waves -> 16(M) x 128(N) tile. K consumed in 64-deep slabs:
// TDM double-buffers A(16x64) and B(128x64) f32 tiles into padded LDS
// (wave 0 issues, S_WAIT_TENSORCNT, workgroup barrier), then each wave
// runs 2x v_wmma_f32_16x16x32_bf16 per slab from conflict-free ds_loads.
// Fragment packing per ISA §7.12.2 (A lanes0-15: K0-7/16-23; B lanes0-15: K0-7/16-23).
// =====================================================================
__global__ __launch_bounds__(256)
void wmma_gemm_xwT(const float* __restrict__ X, const float* __restrict__ W,
                   float* __restrict__ Y, int M, int N, int Kd)
{
    __shared__ __align__(16) float Abuf[2][16 * LROW_];    //  2 x 4.25 KB
    __shared__ __align__(16) float Bbuf[2][128 * LROW_];   //  2 x 34  KB

    const int lane = threadIdx.x & 31;
    const int wave = threadIdx.x >> 5;
    const int m0    = blockIdx.y * 16;
    const int n0blk = blockIdx.x * 128;
    const int n0    = n0blk + wave * 16;
    const int half  = lane >> 4;                 // 0: lanes 0-15, 1: lanes 16-31
    const bool active = (n0 < N);                // wave-uniform (EXEC stays full)
    const unsigned nrows = (unsigned)min(128, N - n0blk);

    const int nslab = Kd / KSLAB_;
    if (threadIdx.x < 32) {                      // wave 0 drives the TDM
        tdm_load_2d((unsigned)(uintptr_t)&Abuf[0][0], X + (size_t)m0 * Kd,
                    KSLAB_, 16u, (unsigned)Kd, (unsigned)M, (unsigned long long)Kd);
        tdm_load_2d((unsigned)(uintptr_t)&Bbuf[0][0], W + (size_t)n0blk * Kd,
                    KSLAB_, nrows, (unsigned)Kd, (unsigned)N, (unsigned long long)Kd);
    }

    v8f acc = {};
    for (int s = 0; s < nslab; ++s) {
        const int cur = s & 1;
        if (threadIdx.x < 32) __builtin_amdgcn_s_wait_tensorcnt(0);
        __syncthreads();                         // slab `cur` visible to all waves
        if (threadIdx.x < 32 && (s + 1) < nslab) {
            const int kk = (s + 1) * KSLAB_;
            tdm_load_2d((unsigned)(uintptr_t)&Abuf[cur ^ 1][0],
                        X + (size_t)m0 * Kd + kk,
                        KSLAB_, 16u, (unsigned)Kd, (unsigned)M, (unsigned long long)Kd);
            tdm_load_2d((unsigned)(uintptr_t)&Bbuf[cur ^ 1][0],
                        W + (size_t)n0blk * Kd + kk,
                        KSLAB_, nrows, (unsigned)Kd, (unsigned)N, (unsigned long long)Kd);
        }
        if (active) {
            #pragma unroll
            for (int kk = 0; kk < KSLAB_; kk += 32) {
                const float* arow = &Abuf[cur][(lane & 15) * LROW_ + kk];
                const float* brow = &Bbuf[cur][(wave * 16 + (lane & 15)) * LROW_ + kk];
                const float4 a0 = *(const float4*)(arow + 8 * half + 0);
                const float4 a1 = *(const float4*)(arow + 8 * half + 4);
                const float4 a2 = *(const float4*)(arow + 16 + 8 * half + 0);
                const float4 a3 = *(const float4*)(arow + 16 + 8 * half + 4);
                const float4 b0 = *(const float4*)(brow + 8 * half + 0);
                const float4 b1 = *(const float4*)(brow + 8 * half + 4);
                const float4 b2 = *(const float4*)(brow + 16 + 8 * half + 0);
                const float4 b3 = *(const float4*)(brow + 16 + 8 * half + 4);
                const float af[16] = {a0.x, a0.y, a0.z, a0.w, a1.x, a1.y, a1.z, a1.w,
                                      a2.x, a2.y, a2.z, a2.w, a3.x, a3.y, a3.z, a3.w};
                const float bf[16] = {b0.x, b0.y, b0.z, b0.w, b1.x, b1.y, b1.z, b1.w,
                                      b2.x, b2.y, b2.z, b2.w, b3.x, b3.y, b3.z, b3.w};
                v16bf a, b;
                #pragma unroll
                for (int i = 0; i < 16; ++i) { a[i] = (__bf16)af[i]; b[i] = (__bf16)bf[i]; }
                acc = __builtin_amdgcn_wmma_f32_16x16x32_bf16(
                          false, a, false, b, (short)0, acc, false, false);
            }
        }
        __syncthreads();                         // all reads of `cur` done
    }

    if (active) {
        #pragma unroll
        for (int r = 0; r < 8; ++r) {
            const int m = m0 + r + 8 * half;
            Y[(size_t)m * N + n0 + (lane & 15)] = acc[r];
        }
    }
}

// =====================================================================
// Kernel 2: causal depthwise conv (K=4, left pad 3) + SiLU
// =====================================================================
__global__ __launch_bounds__(256)
void conv_silu_kernel(const float* __restrict__ lin, const float* __restrict__ cw,
                      float* __restrict__ out)
{
    const int idx = blockIdx.x * 256 + threadIdx.x;      // over T*CONV_DIM
    const int t = idx / CONV_DIM_;
    const int c = idx - t * CONV_DIM_;
    float acc = 0.f;
    #pragma unroll
    for (int j = 0; j < 4; ++j) {
        const int tt = t - 3 + j;
        if (tt >= 0) acc += lin[(size_t)tt * CONV_DIM_ + c] * cw[c * 4 + j];
    }
    out[idx] = silu_(acc);
}

// =====================================================================
// Kernel 3: beta = sigmoid(blin);  g = -exp(A_log)*softplus(alin+dt_bias)
// =====================================================================
__global__ __launch_bounds__(256)
void gate_kernel(const float* __restrict__ blin, const float* __restrict__ alin,
                 const float* __restrict__ dt_bias, const float* __restrict__ A_log,
                 float* __restrict__ beta, float* __restrict__ g, int n)
{
    const int i = blockIdx.x * 256 + threadIdx.x;
    if (i >= n) return;
    const int h = i & (HV_ - 1);
    beta[i] = sigmoid_(blin[i]);
    const float a = alin[i] + dt_bias[h];
    const float sp = (a > 20.f) ? a : log1pf(__expf(a));
    g[i] = -__expf(A_log[h]) * sp;
}

// =====================================================================
// Kernel 4: in-place L2 norm of q / k head rows (DK=128), one wave per row
// =====================================================================
__global__ __launch_bounds__(256)
void l2norm_qk(float* __restrict__ conv, int off, float scale)
{
    const int row = blockIdx.x * 8 + (threadIdx.x >> 5);  // over T*HK rows
    const int lane = threadIdx.x & 31;
    const int t = row >> 4, hk = row & 15;
    float* p = conv + (size_t)t * CONV_DIM_ + off + hk * DK_ + lane * 4;
    float4 v = *(const float4*)p;
    float ss = v.x * v.x + v.y * v.y + v.z * v.z + v.w * v.w;
    #pragma unroll
    for (int o = 16; o > 0; o >>= 1) ss += __shfl_xor(ss, o, 32);
    const float r = rsqrtf(ss + EPS_) * scale;
    v.x *= r; v.y *= r; v.z *= r; v.w *= r;
    *(float4*)p = v;
}

// =====================================================================
// Kernel 5: sequential gated delta-rule scan. One block (256 thr) per head.
// S[128,128] lives in registers: thread (rg=tid>>5, cg=tid&31) owns
// dk in [16*rg,16*rg+16) x dv in [4*cg,4*cg+4)  -> 64 VGPRs of state.
// Cross-wave dot-product reductions go through LDS.
// =====================================================================
__global__ __launch_bounds__(256)
void deltanet_scan(const float* __restrict__ conv, const float* __restrict__ beta,
                   const float* __restrict__ g, float* __restrict__ core)
{
    const int h = blockIdx.x;
    const int hk = h >> 1;                    // GQA repeat_interleave(2)
    const int tid = threadIdx.x;
    const int cg = tid & 31, rg = tid >> 5;
    const int dv0 = cg * 4, dk0 = rg * 16;

    __shared__ float ksh[DK_], qsh[DK_], vsh[DV_];
    __shared__ float red[8][DV_];

    float S[16][4];
    #pragma unroll
    for (int i = 0; i < 16; ++i)
        #pragma unroll
        for (int j = 0; j < 4; ++j) S[i][j] = 0.f;

    for (int t = 0; t < T_; ++t) {
        const float* rowp = conv + (size_t)t * CONV_DIM_;
        if (tid < 128) {
            ksh[tid] = rowp[KEY_DIM_ + hk * DK_ + tid];
            vsh[tid] = rowp[2 * KEY_DIM_ + h * DV_ + tid];
            if (t + 1 < T_)                    // global_prefetch_b8 next row
                __builtin_prefetch(rowp + CONV_DIM_ + KEY_DIM_ + hk * DK_ + tid, 0, 0);
        } else {
            qsh[tid - 128] = rowp[hk * DK_ + (tid - 128)];
        }
        __syncthreads();

        const float dec = __expf(g[t * HV_ + h]);
        const float bt  = beta[t * HV_ + h];

        // decay + partial kv = k . S
        float part[4] = {0.f, 0.f, 0.f, 0.f};
        #pragma unroll
        for (int i = 0; i < 16; ++i) {
            const float kv = ksh[dk0 + i];
            #pragma unroll
            for (int j = 0; j < 4; ++j) { S[i][j] *= dec; part[j] += kv * S[i][j]; }
        }
        #pragma unroll
        for (int j = 0; j < 4; ++j) red[rg][dv0 + j] = part[j];
        __syncthreads();

        float delta[4];
        #pragma unroll
        for (int j = 0; j < 4; ++j) {
            float kvs = 0.f;
            #pragma unroll
            for (int r = 0; r < 8; ++r) kvs += red[r][dv0 + j];
            delta[j] = (vsh[dv0 + j] - kvs) * bt;
        }
        __syncthreads();                       // red consumed before rewrite

        // rank-1 update + partial o = q . S
        float opart[4] = {0.f, 0.f, 0.f, 0.f};
        #pragma unroll
        for (int i = 0; i < 16; ++i) {
            const float kv = ksh[dk0 + i];
            const float qv = qsh[dk0 + i];
            #pragma unroll
            for (int j = 0; j < 4; ++j) { S[i][j] += kv * delta[j]; opart[j] += qv * S[i][j]; }
        }
        #pragma unroll
        for (int j = 0; j < 4; ++j) red[rg][dv0 + j] = opart[j];
        __syncthreads();

        if (rg == 0) {
            #pragma unroll
            for (int j = 0; j < 4; ++j) {
                float o = 0.f;
                #pragma unroll
                for (int r = 0; r < 8; ++r) o += red[r][dv0 + j];
                core[((size_t)t * HV_ + h) * DV_ + dv0 + j] = o;
            }
        }
        __syncthreads();                       // shared tiles reused next step
    }
}

// =====================================================================
// Kernel 6: h = core * silu(z); gated RMSNorm over DV; one wave per row
// =====================================================================
__global__ __launch_bounds__(256)
void gated_rmsnorm(const float* __restrict__ core, const float* __restrict__ z,
                   const float* __restrict__ norm_w, float* __restrict__ out)
{
    const int row = blockIdx.x * 8 + (threadIdx.x >> 5);  // over T*HV rows
    const int lane = threadIdx.x & 31;
    const size_t base = (size_t)row * DV_ + lane * 4;
    const float4 c = *(const float4*)(core + base);
    const float4 zz = *(const float4*)(z + base);
    float4 hh;
    hh.x = c.x * silu_(zz.x); hh.y = c.y * silu_(zz.y);
    hh.z = c.z * silu_(zz.z); hh.w = c.w * silu_(zz.w);
    float ss = hh.x * hh.x + hh.y * hh.y + hh.z * hh.z + hh.w * hh.w;
    #pragma unroll
    for (int o = 16; o > 0; o >>= 1) ss += __shfl_xor(ss, o, 32);
    const float r = rsqrtf(ss * (1.f / DV_) + EPS_);
    const float4 w = *(const float4*)(norm_w + lane * 4);
    float4 res;
    res.x = hh.x * r * w.x; res.y = hh.y * r * w.y;
    res.z = hh.z * r * w.z; res.w = hh.w * r * w.w;
    *(float4*)(out + base) = res;
}

// =====================================================================
// Host orchestration. Workspace layout (floats):
//   mixed 16MB | convb 16MB | z 8MB | core 8MB | hnorm 8MB | small gates
//   total ~56.3 MB (all in d_ws)
// =====================================================================
extern "C" void kernel_launch(void* const* d_in, const int* in_sizes, int n_in,
                              void* d_out, int out_size, void* d_ws, size_t ws_size,
                              hipStream_t stream)
{
    const float* x       = (const float*)d_in[0];
    const float* W_qkv   = (const float*)d_in[1];
    const float* W_z     = (const float*)d_in[2];
    const float* W_b     = (const float*)d_in[3];
    const float* W_a     = (const float*)d_in[4];
    const float* conv_w  = (const float*)d_in[5];
    const float* dt_bias = (const float*)d_in[6];
    const float* A_log   = (const float*)d_in[7];
    const float* norm_w  = (const float*)d_in[8];
    const float* W_out   = (const float*)d_in[9];
    float* out = (float*)d_out;

    float* ws    = (float*)d_ws;
    float* mixed = ws;                                   // [512][8192]
    float* convb = ws + (size_t)4194304;                 // [512][8192]
    float* zbuf  = ws + (size_t)8388608;                 // [512][4096]
    float* core  = ws + (size_t)10485760;                // [512][32][128]
    float* hnorm = ws + (size_t)12582912;                // [512][4096]
    float* blin  = ws + (size_t)14680064;                // [512][32]
    float* alin  = blin + 16384;
    float* beta  = blin + 32768;
    float* gdec  = blin + 49152;

    const dim3 blk(256);

    // input projections (TDM-staged bf16 WMMA, f32 accumulate)
    wmma_gemm_xwT<<<dim3(64, 32), blk, 0, stream>>>(x, W_qkv, mixed, T_, CONV_DIM_, D_);
    wmma_gemm_xwT<<<dim3(32, 32), blk, 0, stream>>>(x, W_z,   zbuf,  T_, VAL_DIM_,  D_);
    wmma_gemm_xwT<<<dim3(1,  32), blk, 0, stream>>>(x, W_b,   blin,  T_, HV_,       D_);
    wmma_gemm_xwT<<<dim3(1,  32), blk, 0, stream>>>(x, W_a,   alin,  T_, HV_,       D_);

    // conv + SiLU, gates
    conv_silu_kernel<<<(T_ * CONV_DIM_) / 256, blk, 0, stream>>>(mixed, conv_w, convb);
    gate_kernel<<<(T_ * HV_) / 256, blk, 0, stream>>>(blin, alin, dt_bias, A_log,
                                                      beta, gdec, T_ * HV_);

    // q/k per-head L2 norm (q also scaled by DK^-1/2); 8 rows per block
    l2norm_qk<<<(T_ * HK_) / 8, blk, 0, stream>>>(convb, 0,        0.0883883476483184f);
    l2norm_qk<<<(T_ * HK_) / 8, blk, 0, stream>>>(convb, KEY_DIM_, 1.0f);

    // sequential recurrence, one WGP-resident block per value head
    deltanet_scan<<<HV_, blk, 0, stream>>>(convb, beta, gdec, core);

    // gated RMSNorm + output projection
    gated_rmsnorm<<<(T_ * HV_) / 8, blk, 0, stream>>>(core, zbuf, norm_w, hnorm);
    wmma_gemm_xwT<<<dim3(16, 32), blk, 0, stream>>>(hnorm, W_out, out, T_, D_, VAL_DIM_);
}
